// MultiHeadAttentionWithRoPE_1168231104642
// MI455X (gfx1250) — matrix operand order repaired and checked by reading, not compiled, not run on hardware
//
#include <hip/hip_runtime.h>
#include <hip/hip_bf16.h>

typedef __attribute__((ext_vector_type(8)))  _Float16 v8h;
typedef __attribute__((ext_vector_type(16))) _Float16 v16h;
typedef __attribute__((ext_vector_type(8)))  float    v8f;

#define D_MODEL 2048
#define N_HEADS 16
#define HEAD_DIM 128
#define BATCH 2
#define SEQ 2048
#define M_TOT (BATCH * SEQ)   // 4096

// ---------------------------------------------------------------------------
// f32 -> f16 conversion
// ---------------------------------------------------------------------------
__global__ void cvt_f32_to_f16(const float* __restrict__ src,
                               _Float16* __restrict__ dst, int n) {
    int i = blockIdx.x * blockDim.x + threadIdx.x;
    if (i < n) dst[i] = (_Float16)src[i];
}

// ---------------------------------------------------------------------------
// C[M,N] = A[M,K](f16,row-major) * W[N,K](f16,row-major)^T + bias
// WG: 256 threads = 8 waves. WG tile 128(M) x 128(N); each wave: 16 x 128.
// B operand (lane = N column, 16 contiguous K per lane-half) staged via LDS.
// ---------------------------------------------------------------------------
template <typename OutT>
__global__ __launch_bounds__(256) void gemm_wmma(
    const _Float16* __restrict__ A, const _Float16* __restrict__ W,
    const float* __restrict__ bias, OutT* __restrict__ C,
    int M, int N, int K) {
    __shared__ __align__(32) _Float16 ldsB[128 * 32];   // [ncol][k]
    const int tid  = threadIdx.x;
    const int wave = tid >> 5;
    const int lane = tid & 31;
    const int hi   = lane >> 4;     // lane half (0/1)
    const int ln   = lane & 15;
    const int mbase = blockIdx.x * 128 + wave * 16;     // wave's 16 rows
    const int nbase = blockIdx.y * 128;

    v8f acc[8];
#pragma unroll
    for (int t = 0; t < 8; ++t)
#pragma unroll
        for (int r = 0; r < 8; ++r) acc[t][r] = 0.0f;

    const _Float16* Arow = A + (size_t)(mbase + ln) * K;
    // cooperative W tile load: thread -> (row 0..127, 16-half segment 0..1)
    const int wrow = tid >> 1;
    const int wseg = tid & 1;
    const _Float16* Wp  = W + (size_t)(nbase + wrow) * K + wseg * 16;
    _Float16*       lWp = &ldsB[wrow * 32 + wseg * 16];

    for (int k0 = 0; k0 < K; k0 += 32) {
        __syncthreads();
        *(v16h*)lWp = *(const v16h*)(Wp + k0);          // 32B global->LDS
        __syncthreads();
        __builtin_prefetch(Wp + k0 + 32, 0, 1);
        __builtin_prefetch(Arow + k0 + 32, 0, 1);

        // A operand: lane-half K pattern {0..15} / {16..31}
        union { v16h v; v8h h2[2]; } a;
        const _Float16* ap = Arow + k0 + hi * 16;
        a.h2[0] = *(const v8h*)(ap);
        a.h2[1] = *(const v8h*)(ap + 8);

#pragma unroll
        for (int t = 0; t < 8; ++t) {
            v16h b = *(const v16h*)&ldsB[(t * 16 + ln) * 32 + hi * 16];
            acc[t] = __builtin_amdgcn_wmma_f32_16x16x32_f16(
                false, a.v, false, b, (short)0, acc[t], false, false);
        }
    }

#pragma unroll
    for (int t = 0; t < 8; ++t) {
        const int col = nbase + t * 16 + ln;
        const float bb = bias[col];
#pragma unroll
        for (int r = 0; r < 8; ++r) {
            const int row = mbase + r + hi * 8;
            C[(size_t)row * N + col] = (OutT)(acc[t][r] + bb);
        }
    }
}

// ---------------------------------------------------------------------------
// RoPE + layout: q,k [B,S,D] -> qr,kr [B,H,S,128] (rotated); v -> vt [B,H,128,S]
// one thread per (b,h,s,i<64) pair
// ---------------------------------------------------------------------------
__global__ void rope_pack(const _Float16* __restrict__ q,
                          const _Float16* __restrict__ k,
                          const _Float16* __restrict__ v,
                          _Float16* __restrict__ qr,
                          _Float16* __restrict__ kr,
                          _Float16* __restrict__ vt) {
    int idx = blockIdx.x * blockDim.x + threadIdx.x;
    if (idx >= BATCH * N_HEADS * SEQ * 64) return;
    const int i = idx & 63;
    const int s = (idx >> 6) & (SEQ - 1);
    const int h = (idx >> 17) & (N_HEADS - 1);
    const int b = idx >> 21;

    const float inv = __powf(10000.0f, -(float)(2 * i) / 128.0f);
    const float f = (float)s * inv;
    const float c = __cosf(f), sn = __sinf(f);

    const size_t src = ((size_t)(b * SEQ + s)) * D_MODEL + h * HEAD_DIM;
    const float q1 = (float)q[src + i], q2 = (float)q[src + i + 64];
    const float k1 = (float)k[src + i], k2 = (float)k[src + i + 64];

    const size_t dst = (((size_t)(b * N_HEADS + h)) * SEQ + s) * HEAD_DIM;
    qr[dst + i]      = (_Float16)(q1 * c - q2 * sn);
    qr[dst + i + 64] = (_Float16)(q2 * c + q1 * sn);
    kr[dst + i]      = (_Float16)(k1 * c - k2 * sn);
    kr[dst + i + 64] = (_Float16)(k2 * c + k1 * sn);

    const size_t vh = (size_t)(b * N_HEADS + h) * HEAD_DIM;
    vt[(vh + i) * SEQ + s]      = v[src + i];
    vt[(vh + i + 64) * SEQ + s] = v[src + i + 64];
}

// ---------------------------------------------------------------------------
// Flash-style causal attention. Grid: (S/128, H, B). 8 waves; wave owns 16 q rows.
// scores = Q*K^T (4 wmma per 16-key tile), online softmax, P via per-wave LDS
// staged into A-operand layout, out += P*V (8 wmma per 32-key block).
// ---------------------------------------------------------------------------
__global__ __launch_bounds__(256) void attn_wmma(
    const _Float16* __restrict__ qr, const _Float16* __restrict__ kr,
    const _Float16* __restrict__ vt, _Float16* __restrict__ ao) {
    __shared__ __align__(32) _Float16 ldsP[8 * 16 * 32];   // per-wave 16x32 P tile
    const int tid  = threadIdx.x;
    const int wave = tid >> 5;
    const int lane = tid & 31;
    const int hi   = lane >> 4;
    const int ln   = lane & 15;
    const int qb = blockIdx.x;          // 0..15
    const int h  = blockIdx.y;
    const int b  = blockIdx.z;

    const size_t head = (size_t)(b * N_HEADS + h) * SEQ * HEAD_DIM;
    const _Float16* Q  = qr + head;
    const _Float16* Kp = kr + head;
    const _Float16* Vt = vt + head;     // [128][SEQ]
    const int qbase = qb * 128 + wave * 16;

    // Q A-operand tiles: 4 chunks of K(d)=32
    union { v16h v; v8h h2[2]; } aq[4];
    const _Float16* qrow = Q + (size_t)(qbase + ln) * HEAD_DIM;
#pragma unroll
    for (int c = 0; c < 4; ++c) {
        const _Float16* p = qrow + c * 32 + hi * 16;
        aq[c].h2[0] = *(const v8h*)(p);
        aq[c].h2[1] = *(const v8h*)(p + 8);
    }

    float mrow[8], lrow[8];
    v8f acc[8];
#pragma unroll
    for (int r = 0; r < 8; ++r) { mrow[r] = -1e30f; lrow[r] = 0.0f; }
#pragma unroll
    for (int t = 0; t < 8; ++t)
#pragma unroll
        for (int r = 0; r < 8; ++r) acc[t][r] = 0.0f;

    const float scale = 0.08838834764831845f;   // 1/sqrt(128)
    _Float16* lp = &ldsP[wave * 512];
    const int nblocks = (qbase + 15) / 32 + 1;  // causal 32-key blocks

    for (int jb = 0; jb < nblocks; ++jb) {
        const int kbase = jb * 32;
        v8f st[2];
#pragma unroll
        for (int u = 0; u < 2; ++u) {
#pragma unroll
            for (int r = 0; r < 8; ++r) st[u][r] = 0.0f;
            const _Float16* krow = Kp + (size_t)(kbase + u * 16 + ln) * HEAD_DIM;
            __builtin_prefetch(krow + 64 * HEAD_DIM, 0, 1);
#pragma unroll
            for (int c = 0; c < 4; ++c) {
                v16h bk = *(const v16h*)(krow + c * 32 + hi * 16);
                st[u] = __builtin_amdgcn_wmma_f32_16x16x32_f16(
                    false, aq[c].v, false, bk, (short)0, st[u], false, false);
            }
            // scale + causal mask
#pragma unroll
            for (int r = 0; r < 8; ++r) {
                const int qrow_g = qbase + r + hi * 8;
                const int key = kbase + u * 16 + ln;
                st[u][r] = (key <= qrow_g) ? st[u][r] * scale : -1e30f;
            }
        }
        // online softmax (row = 16 lanes within a half; masks 1..8 stay in-half)
#pragma unroll
        for (int r = 0; r < 8; ++r) {
            float mx = fmaxf(st[0][r], st[1][r]);
#pragma unroll
            for (int msk = 8; msk >= 1; msk >>= 1)
                mx = fmaxf(mx, __shfl_xor(mx, msk));
            const float mnew = fmaxf(mrow[r], mx);
            const float sc = __expf(mrow[r] - mnew);
            mrow[r] = mnew;
            const float p0 = __expf(st[0][r] - mnew);
            const float p1 = __expf(st[1][r] - mnew);
            st[0][r] = p0; st[1][r] = p1;
            float rs = p0 + p1;
#pragma unroll
            for (int msk = 8; msk >= 1; msk >>= 1)
                rs += __shfl_xor(rs, msk);
            lrow[r] = lrow[r] * sc + rs;
#pragma unroll
            for (int t = 0; t < 8; ++t) acc[t][r] *= sc;
        }
        // P -> LDS in A-operand layout: pos = col (contiguous K per lane half)
#pragma unroll
        for (int u = 0; u < 2; ++u) {
            const int col = u * 16 + ln;
            const int pos = col;
#pragma unroll
            for (int r = 0; r < 8; ++r)
                lp[(r + hi * 8) * 32 + pos] = (_Float16)st[u][r];
        }
        const v16h pa = *(const v16h*)&lp[ln * 32 + hi * 16];
        // out += P * V
#pragma unroll
        for (int t = 0; t < 8; ++t) {
            const v16h bv = *(const v16h*)(Vt + (size_t)(t * 16 + ln) * SEQ +
                                           kbase + hi * 16);
            acc[t] = __builtin_amdgcn_wmma_f32_16x16x32_f16(
                false, pa, false, bv, (short)0, acc[t], false, false);
        }
    }
    // normalize + write [B,S,D] f16
#pragma unroll
    for (int t = 0; t < 8; ++t) {
        const int col = h * HEAD_DIM + t * 16 + ln;
#pragma unroll
        for (int r = 0; r < 8; ++r) {
            const int s = qbase + r + hi * 8;
            ao[((size_t)(b * SEQ + s)) * D_MODEL + col] =
                (_Float16)(acc[t][r] / lrow[r]);
        }
    }
}

// ---------------------------------------------------------------------------
// host-side launcher
// ---------------------------------------------------------------------------
extern "C" void kernel_launch(void* const* d_in, const int* in_sizes, int n_in,
                              void* d_out, int out_size, void* d_ws, size_t ws_size,
                              hipStream_t stream) {
    const float* x  = (const float*)d_in[0];
    const float* wq = (const float*)d_in[1];
    const float* bq = (const float*)d_in[2];
    const float* wk = (const float*)d_in[3];
    const float* bk = (const float*)d_in[4];
    const float* wv = (const float*)d_in[5];
    const float* bv = (const float*)d_in[6];
    const float* wo = (const float*)d_in[7];
    const float* bo = (const float*)d_in[8];

    _Float16* ws = (_Float16*)d_ws;
    const size_t NX = (size_t)M_TOT * D_MODEL;        // 8,388,608
    const size_t NW = (size_t)D_MODEL * D_MODEL;      // 4,194,304
    _Float16* XH = ws;
    _Float16* WQ = XH + NX;
    _Float16* WK = WQ + NW;
    _Float16* WV = WK + NW;
    _Float16* WO = WV + NW;
    _Float16* QH = WO + NW;
    _Float16* KH = QH + NX;
    _Float16* VH = KH + NX;
    _Float16* QR = VH + NX;
    _Float16* KR = QR + NX;
    _Float16* VT = KR + NX;
    _Float16* AO = QH;   // alias: Q-projection dead after rope_pack

    const int thr = 256;
    cvt_f32_to_f16<<<(int)((NX + thr - 1) / thr), thr, 0, stream>>>(x, XH, (int)NX);
    cvt_f32_to_f16<<<(int)((NW + thr - 1) / thr), thr, 0, stream>>>(wq, WQ, (int)NW);
    cvt_f32_to_f16<<<(int)((NW + thr - 1) / thr), thr, 0, stream>>>(wk, WK, (int)NW);
    cvt_f32_to_f16<<<(int)((NW + thr - 1) / thr), thr, 0, stream>>>(wv, WV, (int)NW);
    cvt_f32_to_f16<<<(int)((NW + thr - 1) / thr), thr, 0, stream>>>(wo, WO, (int)NW);

    dim3 ggrid(M_TOT / 128, D_MODEL / 128);
    gemm_wmma<_Float16><<<ggrid, 256, 0, stream>>>(XH, WQ, bq, QH, M_TOT, D_MODEL, D_MODEL);
    gemm_wmma<_Float16><<<ggrid, 256, 0, stream>>>(XH, WK, bk, KH, M_TOT, D_MODEL, D_MODEL);
    gemm_wmma<_Float16><<<ggrid, 256, 0, stream>>>(XH, WV, bv, VH, M_TOT, D_MODEL, D_MODEL);

    const int nrope = BATCH * N_HEADS * SEQ * 64;
    rope_pack<<<(nrope + thr - 1) / thr, thr, 0, stream>>>(QH, KH, VH, QR, KR, VT);

    attn_wmma<<<dim3(SEQ / 128, N_HEADS, BATCH), 256, 0, stream>>>(QR, KR, VT, AO);

    gemm_wmma<float><<<ggrid, 256, 0, stream>>>(AO, WO, bo, (float*)d_out,
                                                M_TOT, D_MODEL, D_MODEL);
}
